// ArcHybridLSTMModel_26079041421718
// MI455X (gfx1250) — compile-verified, matrix operand order checked by reading, not run on hardware
//
#include <hip/hip_runtime.h>
#include <math.h>

// ---------------------------------------------------------------------------
// ArcHybridLSTM forward for MI455X (gfx1250, wave32, WMMA bf16).
//
// Pipeline:
//   embed -> (xg GEMM, lstm recurrence) x2 layers -> fused MLP+head x2
// Large matmuls: v_wmma_f32_16x16x32_bf16. GEMM staging uses
// global_load_async_to_lds (ASYNCcnt) with double buffering.
// ---------------------------------------------------------------------------

typedef __attribute__((ext_vector_type(16))) __bf16 v16bf;
typedef __attribute__((ext_vector_type(8)))  __bf16 v8bf;
typedef __attribute__((ext_vector_type(4)))  __bf16 v4bf;
typedef __attribute__((ext_vector_type(8)))  float  v8f;
typedef int v4i_ __attribute__((vector_size(16)));
typedef int v2i_ __attribute__((vector_size(8)));

#define B_    64
#define S_    256
#define T_    256
#define H_    256
#define G4H   1024
#define FEAT_ 6144
#define HID_  512
#define MTOT  16384   // B*S = B*T
#define SLOTS 12

#if __has_builtin(__builtin_amdgcn_global_load_async_to_lds_b128) && \
    __has_builtin(__builtin_amdgcn_global_load_async_to_lds_b64) &&  \
    __has_builtin(__builtin_amdgcn_s_wait_asynccnt)
#define HAVE_ASYNC_LDS 1
#else
#define HAVE_ASYNC_LDS 0
#endif

// 16-byte global -> LDS copy (async; tracked by ASYNCcnt).
__device__ __forceinline__ void cp16(const __bf16* g, __bf16* l) {
#if HAVE_ASYNC_LDS
  __builtin_amdgcn_global_load_async_to_lds_b128(
      (__attribute__((address_space(1))) v4i_*)(const void*)g,
      (__attribute__((address_space(3))) v4i_*)(void*)l, 0, 0);
#else
  *(v8bf*)l = *(const v8bf*)g;
#endif
}
// 8-byte global -> LDS copy.
__device__ __forceinline__ void cp8(const __bf16* g, __bf16* l) {
#if HAVE_ASYNC_LDS
  __builtin_amdgcn_global_load_async_to_lds_b64(
      (__attribute__((address_space(1))) v2i_*)(const void*)g,
      (__attribute__((address_space(3))) v2i_*)(void*)l, 0, 0);
#else
  *(v4bf*)l = *(const v4bf*)g;
#endif
}
template <int N>
__device__ __forceinline__ void wait_async() {
#if HAVE_ASYNC_LDS
  __builtin_amdgcn_s_wait_asynccnt(N);
#endif
}

__device__ __forceinline__ v8f wmma_bf(v16bf a, v16bf b, v8f c) {
  // v_wmma_f32_16x16x32_bf16: D = A(16x32) * B(32x16) + C(16x16 f32)
  return __builtin_amdgcn_wmma_f32_16x16x32_bf16(false, a, false, b,
                                                 (short)0, c, false, false);
}

// Load a 16x32 bf16 fragment (A layout; B uses identical n-major layout since
// both GEMMs are X @ W^T with W stored [N,K] row-major).
// ISA 7.12.2: lanes 0-15 hold k = 0..7,16..23 ; lanes 16-31 hold k = 8..15,24..31.
__device__ __forceinline__ v16bf load_frag(const __bf16* __restrict__ base,
                                           int row0, int stride, int kbase) {
  const int lane = threadIdx.x & 31;
  const int r    = lane & 15;
  const int kofs = (lane >> 4) * 8;
  const __bf16* p = base + (size_t)(row0 + r) * stride + kbase + kofs;
  v8bf lo = *(const v8bf*)p;
  v8bf hi = *(const v8bf*)(p + 16);
  v16bf o;
#pragma unroll
  for (int i = 0; i < 8; ++i) { o[i] = lo[i]; o[i + 8] = hi[i]; }
  return o;
}

// ---------------------------------------------------------------------------
// Utility kernels
// ---------------------------------------------------------------------------
__global__ void cvt_bf16_kernel(const float* __restrict__ src,
                                __bf16* __restrict__ dst, int n) {
  int i = blockIdx.x * 256 + threadIdx.x;
  if (i < n) dst[i] = (__bf16)src[i];
}

__global__ void bias_sum_kernel(const float* __restrict__ a,
                                const float* __restrict__ b,
                                float* __restrict__ o, int n) {
  int i = blockIdx.x * 256 + threadIdx.x;
  if (i < n) o[i] = a[i] + b[i];
}

// ivec = cat(wlookup[word_ids], plookup[pos_ids]) -> bf16 [MTOT,160]
__global__ void embed_kernel(const int* __restrict__ wid,
                             const int* __restrict__ pid,
                             const float* __restrict__ wl,
                             const float* __restrict__ pl,
                             __bf16* __restrict__ ivec) {
  int i = blockIdx.x * 256 + threadIdx.x;
  if (i >= MTOT * 160) return;
  int tok = i / 160, d = i - tok * 160;
  float v = (d < 128) ? wl[(size_t)wid[tok] * 128 + d]
                      : pl[(size_t)pid[tok] * 32 + (d - 128)];
  ivec[i] = (__bf16)v;
}

// ---------------------------------------------------------------------------
// xg GEMM:  xg[M,1024] = A[M,K](bf16) @ W[1024,K](bf16)^T + biassum
// Block tile 64x256 (8 waves, wave tile 16x128), double-buffered async staging.
// grid = (1024/256, MTOT/64) = (4, 256).  5 async loads / thread / stage.
// ---------------------------------------------------------------------------
__global__ __launch_bounds__(256) void gemm_xg_kernel(
    const __bf16* __restrict__ A, int K,
    const __bf16* __restrict__ W,
    const float* __restrict__ bias,
    float* __restrict__ xg) {
  __shared__ __bf16 sA[2][64 * 32];    //  2 x 4 KB
  __shared__ __bf16 sB[2][256 * 32];   //  2 x 16 KB
  const int tid = threadIdx.x, wid = tid >> 5, lane = tid & 31;
  const int m0 = blockIdx.y * 64;
  const int n0 = blockIdx.x * 256;
  const int wm = (wid & 3) * 16;
  const int wn = (wid >> 2) * 128;

  auto stage = [&](int buf, int k0) {
    int row = tid >> 2, col = (tid & 3) * 8;
    cp16(&A[(size_t)(m0 + row) * K + k0 + col], &sA[buf][row * 32 + col]);
#pragma unroll
    for (int r = 0; r < 4; ++r) {
      int rw = row + r * 64;
      cp16(&W[(size_t)(n0 + rw) * K + k0 + col], &sB[buf][rw * 32 + col]);
    }
  };

  v8f acc[8] = {};
  const int nK = K >> 5;
  stage(0, 0);
  for (int ks = 0; ks < nK; ++ks) {
    const int cur = ks & 1;
    if (ks + 1 < nK) {
      stage(cur ^ 1, (ks + 1) << 5);
      wait_async<5>();   // retire current stage; next stays in flight
    } else {
      wait_async<0>();
    }
    __syncthreads();
    v16bf a = load_frag(&sA[cur][0], wm, 32, 0);
#pragma unroll
    for (int t = 0; t < 8; ++t) {
      v16bf b = load_frag(&sB[cur][0], wn + t * 16, 32, 0);
      acc[t] = wmma_bf(a, b, acc[t]);
    }
    __syncthreads();
  }
  const int nl = lane & 15, mb = (lane >> 4) * 8;
#pragma unroll
  for (int t = 0; t < 8; ++t)
#pragma unroll
    for (int e = 0; e < 8; ++e) {
      int m = m0 + wm + mb + e;
      int n = n0 + wn + t * 16 + nl;
      xg[(size_t)m * G4H + n] = acc[t][e] + bias[n];
    }
}

// ---------------------------------------------------------------------------
// LSTM recurrence. Batch split across 4 WGs (16 rows each), grid.y = direction.
// Per step: g = h(16x256) @ Whh^T(256x1024) via WMMA; wave w owns hidden cols
// [w*32, w*32+32) across ALL FOUR gates (tile t = gate*2+half) so i/f/g/o for
// a (row,col) stay in one lane's registers — no gate LDS exchange.
// c in registers; h tile (16x256 bf16, 8 KB) in LDS; Whh (512 KB bf16) from L2.
// ---------------------------------------------------------------------------
__global__ __launch_bounds__(256) void lstm_kernel(
    const float* __restrict__ xg_f, const float* __restrict__ xg_b,
    const __bf16* __restrict__ Whh_f, const __bf16* __restrict__ Whh_b,
    __bf16* __restrict__ vec_out /* [B,S,512] bf16 */) {
  __shared__ __bf16 h_lds[16 * H_];
  const int tid = threadIdx.x, wid = tid >> 5, lane = tid & 31;
  const int b0 = blockIdx.x * 16;
  const bool rev = (blockIdx.y == 1);
  const float* xg = rev ? xg_b : xg_f;
  const __bf16* Whh = rev ? Whh_b : Whh_f;
  const int col_ofs = rev ? H_ : 0;
  for (int i = tid; i < 16 * H_; i += 256) h_lds[i] = (__bf16)0.0f;
  float c_reg[2][8];
#pragma unroll
  for (int h = 0; h < 2; ++h)
#pragma unroll
    for (int e = 0; e < 8; ++e) c_reg[h][e] = 0.0f;
  const int nl = lane & 15, mb = (lane >> 4) * 8;
  __syncthreads();

  for (int s = 0; s < S_; ++s) {
    const int pos = rev ? (S_ - 1 - s) : s;
    v8f acc[8] = {};
    for (int kk = 0; kk < H_; kk += 32) {
      v16bf a = load_frag(h_lds, 0, H_, kk);
#pragma unroll
      for (int t = 0; t < 8; ++t) {
        const int gate = t >> 1, half = t & 1;
        const int nrow = gate * H_ + wid * 32 + half * 16;
        v16bf b = load_frag(Whh, nrow, H_, kk);
        acc[t] = wmma_bf(a, b, acc[t]);
      }
    }
    __syncthreads();  // all h_lds reads complete before overwrite
#pragma unroll
    for (int half = 0; half < 2; ++half) {
      const int col = wid * 32 + half * 16 + nl;
#pragma unroll
      for (int e = 0; e < 8; ++e) {
        const int m = mb + e;
        const size_t xb =
            ((size_t)(b0 + m) * S_ + pos) * G4H + wid * 32 + half * 16 + nl;
        float gi = acc[0 + half][e] + xg[xb + 0 * H_];
        float gf = acc[2 + half][e] + xg[xb + 1 * H_];
        float gg = acc[4 + half][e] + xg[xb + 2 * H_];
        float go = acc[6 + half][e] + xg[xb + 3 * H_];
        float iv = 1.0f / (1.0f + __expf(-gi));
        float fv = 1.0f / (1.0f + __expf(-gf));
        float gv = tanhf(gg);
        float ov = 1.0f / (1.0f + __expf(-go));
        float c = fv * c_reg[half][e] + iv * gv;
        c_reg[half][e] = c;
        float hv = ov * tanhf(c);
        h_lds[m * H_ + col] = (__bf16)hv;
        vec_out[((size_t)(b0 + m) * S_ + pos) * 512 + col_ofs + col] = (__bf16)hv;
      }
    }
    __syncthreads();
  }
}

// ---------------------------------------------------------------------------
// Fused MLP + head:
//   hid = tanh( gather(vec, feat_idx)[32,6144] @ hidW[512,6144]^T + bias )
//   out[32,R]  = hid @ outW[R,512]^T + outBias
// A-tiles gathered on the fly (feats never materialized; vec is 16 MB, L2-hot).
// Block tile 32x512 (8 waves, wave tile 16x128); double-buffered async staging
// (9 async loads / thread / stage); sHid reuses the staging arena after K-loop.
// grid = MTOT/32 = 512
// ---------------------------------------------------------------------------
#define MLP_BUFB 34816  // bytes per staging buffer: sA 2 KB + sB 32 KB

__global__ __launch_bounds__(256) void mlp_kernel(
    const __bf16* __restrict__ vec,      // [B,S,512] bf16
    const int* __restrict__ feat_idx,    // [B,T,12]
    const __bf16* __restrict__ hidW,     // [512,6144] bf16
    const float* __restrict__ hidBias,   // [512]
    const float* __restrict__ outW,      // [R,512] f32
    const float* __restrict__ outBias,   // [R]
    float* __restrict__ out,             // [MTOT,R]
    int R) {
  // Arena: 2 x (sA 2KB + sB 32KB) = 68 KB; sHid (64 KB) aliases it afterwards.
  __shared__ __align__(16) char smem[2 * MLP_BUFB];

  const int tid = threadIdx.x, wid = tid >> 5, lane = tid & 31;
  const int m0 = blockIdx.x * 32;
  const int wm = (wid & 1) * 16;
  const int wn = (wid >> 1) * 128;

  // Runtime-computed LDS buffer addresses (no constant-initializer tables).
  auto sAp = [&](int buf) -> __bf16* { return (__bf16*)(smem + MLP_BUFB * buf); };
  auto sBp = [&](int buf) -> __bf16* { return (__bf16*)(smem + MLP_BUFB * buf + 2048); };

  auto stage = [&](int buf, int k0) {
    const int slot = k0 >> 9;  // k0/512; a 32-wide k-step never straddles slots
    const int d0 = k0 & 511;
    {  // gather A: 32 rows x 32 halves, 8B per thread
      int row = tid >> 3, col = (tid & 7) * 4;
      int m = m0 + row;
      int b = m >> 8;
      int idx = feat_idx[(size_t)m * SLOTS + slot];
      cp8(&vec[((size_t)(b << 8) + idx) * 512 + d0 + col],
          sAp(buf) + row * 32 + col);
    }
    {  // stage B: 512 rows x 32 halves
      int col = (tid & 3) * 8, rbase = tid >> 2;
#pragma unroll
      for (int r = 0; r < 8; ++r) {
        int row = rbase + r * 64;
        cp16(&hidW[(size_t)row * FEAT_ + k0 + col], sBp(buf) + row * 32 + col);
      }
    }
  };

  v8f acc[8] = {};
  const int nK = FEAT_ >> 5;  // 192
  stage(0, 0);
  for (int ks = 0; ks < nK; ++ks) {
    const int cur = ks & 1;
    if (ks + 1 < nK) {
      stage(cur ^ 1, (ks + 1) << 5);
      wait_async<9>();
    } else {
      wait_async<0>();
    }
    __syncthreads();
    v16bf a = load_frag(sAp(cur), wm, 32, 0);
#pragma unroll
    for (int t = 0; t < 8; ++t) {
      v16bf b = load_frag(sBp(cur), wn + t * 16, 32, 0);
      acc[t] = wmma_bf(a, b, acc[t]);
    }
    __syncthreads();
  }
  // tanh + bias into sHid (aliases staging; safe: barrier fenced all reads)
  float* sHid = (float*)(smem + 0);  // 32*512*4 = 64 KB
  const int nl = lane & 15, mb = (lane >> 4) * 8;
#pragma unroll
  for (int t = 0; t < 8; ++t)
#pragma unroll
    for (int e = 0; e < 8; ++e) {
      int m = wm + mb + e;
      int n = wn + t * 16 + nl;
      sHid[m * HID_ + n] = tanhf(acc[t][e] + hidBias[n]);
    }
  __syncthreads();
  // tiny head: 32 x R dot-products of length 512 (outW is L2-hot)
  for (int e = tid; e < 32 * R; e += 256) {
    int row = e / R, o = e - row * R;
    float dot = outBias[o];
    const float* hw = &sHid[row * HID_];
    const float* ww = &outW[(size_t)o * HID_];
#pragma unroll 4
    for (int k = 0; k < HID_; ++k) dot += hw[k] * ww[k];
    out[(size_t)(m0 + row) * R + o] = dot;
  }
}

// ---------------------------------------------------------------------------
// Host launcher
// ---------------------------------------------------------------------------
extern "C" void kernel_launch(void* const* d_in, const int* in_sizes, int n_in,
                              void* d_out, int out_size, void* d_ws,
                              size_t ws_size, hipStream_t stream) {
  const int* word_ids = (const int*)d_in[0];
  const int* pos_ids  = (const int*)d_in[1];
  const int* feat_idx = (const int*)d_in[2];
  const float* wlookup = (const float*)d_in[3];
  const float* plookup = (const float*)d_in[4];
  const float* Wih[4]; const float* Whh[4];
  const float* bih[4]; const float* bhh[4];
  for (int i = 0; i < 4; ++i) {
    Wih[i] = (const float*)d_in[5 + 4 * i + 0];
    Whh[i] = (const float*)d_in[5 + 4 * i + 1];
    bih[i] = (const float*)d_in[5 + 4 * i + 2];
    bhh[i] = (const float*)d_in[5 + 4 * i + 3];
  }
  const float* hidLayer = (const float*)d_in[21];
  const float* hidBias  = (const float*)d_in[22];
  const float* outLayer = (const float*)d_in[23];
  const float* outBias  = (const float*)d_in[24];
  const float* rhidLayer = (const float*)d_in[25];
  const float* rhidBias  = (const float*)d_in[26];
  const float* routLayer = (const float*)d_in[27];
  const float* routBias  = (const float*)d_in[28];

  char* ws = (char*)d_ws;
  size_t off = 0;
  auto take = [&](size_t bytes) -> char* {
    off = (off + 255) & ~(size_t)255;
    char* p = ws + off;
    off += bytes;
    return p;
  };
  __bf16* ivec_b  = (__bf16*)take((size_t)MTOT * 160 * 2);
  const int wihK[4] = {160, 160, 512, 512};
  __bf16* wih_b[4]; __bf16* whh_b[4]; float* bsum[4];
  for (int i = 0; i < 4; ++i) {
    wih_b[i] = (__bf16*)take((size_t)G4H * wihK[i] * 2);
    whh_b[i] = (__bf16*)take((size_t)G4H * H_ * 2);
    bsum[i]  = (float*)take((size_t)G4H * 4);
  }
  float* xg_f = (float*)take((size_t)MTOT * G4H * 4);
  float* xg_b = (float*)take((size_t)MTOT * G4H * 4);
  __bf16* vec1_b = (__bf16*)take((size_t)MTOT * 512 * 2);
  __bf16* vec2_b = (__bf16*)take((size_t)MTOT * 512 * 2);
  __bf16* hidW_b  = (__bf16*)take((size_t)HID_ * FEAT_ * 2);
  __bf16* rhidW_b = (__bf16*)take((size_t)HID_ * FEAT_ * 2);

  auto cvt = [&](const float* s, __bf16* d, int n) {
    cvt_bf16_kernel<<<dim3((n + 255) / 256), dim3(256), 0, stream>>>(s, d, n);
  };
  for (int i = 0; i < 4; ++i) {
    cvt(Wih[i], wih_b[i], G4H * wihK[i]);
    cvt(Whh[i], whh_b[i], G4H * H_);
    bias_sum_kernel<<<dim3(4), dim3(256), 0, stream>>>(bih[i], bhh[i], bsum[i], G4H);
  }
  cvt(hidLayer, hidW_b, HID_ * FEAT_);
  cvt(rhidLayer, rhidW_b, HID_ * FEAT_);

  embed_kernel<<<dim3((MTOT * 160 + 255) / 256), dim3(256), 0, stream>>>(
      word_ids, pos_ids, wlookup, plookup, ivec_b);

  const dim3 gXG(G4H / 256, MTOT / 64);
  // Layer 1
  gemm_xg_kernel<<<gXG, 256, 0, stream>>>(ivec_b, 160, wih_b[0], bsum[0], xg_f);
  gemm_xg_kernel<<<gXG, 256, 0, stream>>>(ivec_b, 160, wih_b[1], bsum[1], xg_b);
  lstm_kernel<<<dim3(4, 2), 256, 0, stream>>>(xg_f, xg_b, whh_b[0], whh_b[1], vec1_b);
  // Layer 2
  gemm_xg_kernel<<<gXG, 256, 0, stream>>>(vec1_b, 512, wih_b[2], bsum[2], xg_f);
  gemm_xg_kernel<<<gXG, 256, 0, stream>>>(vec1_b, 512, wih_b[3], bsum[3], xg_b);
  lstm_kernel<<<dim3(4, 2), 256, 0, stream>>>(xg_f, xg_b, whh_b[2], whh_b[3], vec2_b);
  // Fused MLP heads
  float* out0 = (float*)d_out;
  float* rout0 = out0 + (size_t)MTOT * 3;
  mlp_kernel<<<dim3(MTOT / 32), 256, 0, stream>>>(
      vec2_b, feat_idx, hidW_b, hidBias, outLayer, outBias, out0, 3);
  mlp_kernel<<<dim3(MTOT / 32), 256, 0, stream>>>(
      vec2_b, feat_idx, rhidW_b, rhidBias, routLayer, routBias, rout0, 93);
}
